// GaussianQueryAssignment_49194555408941
// MI455X (gfx1250) — compile-verified, matrix operand order checked by reading, not run on hardware
//
#include <hip/hip_runtime.h>
#include <hip/hip_bf16.h>
#include <math.h>

// ---------------------------------------------------------------------------
// GaussianQueryAssignment for MI455X (gfx1250, wave32, WMMA)
// B=2 T=4 N=20000 M=128; output (2, 80000, 129) f32.
// GEMMs (gf = gfeat@WgT, sim = gf@qfT) run on v_wmma_f32_16x16x32_f16.
// qf tile staged into LDS with global_load_async_to_lds_b128 (ASYNCcnt path).
// ---------------------------------------------------------------------------

typedef __attribute__((ext_vector_type(16))) _Float16 v16h;
typedef __attribute__((ext_vector_type(8)))  _Float16 v8h;
typedef __attribute__((ext_vector_type(8)))  float    v8f;

#define B_ 2
#define T_ 4
#define N_ 20000
#define M_ 128
#define DQ_ 256          // query_feat dim
#define D_  128          // projected dim
#define G_  48           // gfeat dim (32 affinity + 16 motion)
#define GP_ 64           // gfeat dim padded for K
#define ITILE 64         // instances per block
#define NTILES ((N_ + ITILE - 1) / ITILE)
#define BT_ (B_ * T_)

#define DILATION  1.5f
#define MIN_EXT   0.5f
#define GATE_TEMP 3.0f
#define FEAT_TEMP 6.0f
#define KEEP_W    0.5f

// Load a v16h WMMA fragment (A or B role) from a row-major f16 row in LDS.
// 16-bit operand layout: lane L (row/col = L%16) holds K-chunks
// [kbase+klo .. +7] and [kbase+16+klo .. +7], klo = (L>=16)?8:0.
__device__ __forceinline__ v16h load_frag16(const _Float16* row, int kbase, int klo) {
    union { v16h v; v8h h[2]; } u;
    u.h[0] = *(const v8h*)(row + kbase + klo);
    u.h[1] = *(const v8h*)(row + kbase + 16 + klo);
    return u.v;
}

// ---------------------------------------------------------------------------
// Kernel 1: per (b,t) compute qf = l2norm(query_feat @ Wq.T + bq)  -> f16,
// plus per-query metadata {cx,cy,cz, invx,invy,invz, presence, 0}.
// grid = BT_, block = 256
// ---------------------------------------------------------------------------
__global__ void gqa_prep_kernel(const float* __restrict__ qbox,
                                const float* __restrict__ qfeat,
                                const float* __restrict__ pres,
                                const float* __restrict__ Wq,
                                const float* __restrict__ bq,
                                _Float16*    __restrict__ qf_h,
                                float*       __restrict__ qmeta,
                                float*       __restrict__ qf_raw) {
    const int bt  = blockIdx.x;
    const int tid = threadIdx.x;

    __shared__ float ssq[M_];
    if (tid < M_) ssq[tid] = 0.0f;

    if (tid < M_) {
        const float* bx = qbox + (size_t)(bt * M_ + tid) * 7;
        float sx = fmaxf(fabsf(bx[3]), MIN_EXT);
        float sy = fmaxf(fabsf(bx[4]), MIN_EXT);
        float sz = fmaxf(fabsf(bx[5]), MIN_EXT);
        float* qm = qmeta + (size_t)(bt * M_ + tid) * 8;
        qm[0] = bx[0]; qm[1] = bx[1]; qm[2] = bx[2];
        qm[3] = 1.0f / (0.5f * DILATION * sx);
        qm[4] = 1.0f / (0.5f * DILATION * sy);
        qm[5] = 1.0f / (0.5f * DILATION * sz);
        qm[6] = pres[bt * M_ + tid];
        qm[7] = 0.0f;
    }
    __syncthreads();

    const float* qF  = qfeat  + (size_t)bt * M_ * DQ_;
    float*       raw = qf_raw + (size_t)bt * M_ * D_;

    for (int od = tid; od < M_ * D_; od += 256) {
        int m = od >> 7, d = od & (D_ - 1);
        const float4* x = (const float4*)(qF + (size_t)m * DQ_);
        const float4* w = (const float4*)(Wq + (size_t)d * DQ_);
        float s = bq[d];
        #pragma unroll 8
        for (int k = 0; k < DQ_ / 4; ++k) {
            float4 xv = x[k], wv = w[k];
            s += xv.x * wv.x + xv.y * wv.y + xv.z * wv.z + xv.w * wv.w;
        }
        raw[od] = s;
        atomicAdd(&ssq[m], s * s);
    }
    __syncthreads();

    for (int od = tid; od < M_ * D_; od += 256) {
        int m = od >> 7;
        float scl = 1.0f / fmaxf(sqrtf(ssq[m]), 1e-12f);
        qf_h[(size_t)bt * M_ * D_ + od] = (_Float16)(raw[od] * scl);
    }
}

// ---------------------------------------------------------------------------
// Kernel 2: main fused kernel. grid = (NTILES, BT_), block = 128 (4 waves).
// Each block: 64 instances x 128 queries.
// ---------------------------------------------------------------------------
struct Phase1LDS {
    _Float16 g[ITILE * GP_];   // gfeat tile, K padded to 64   ( 8 KB)
    _Float16 wg[D_ * GP_];     // Wg (row d, K padded)         (16 KB)
};
union SmemAB {
    Phase1LDS p1;
    _Float16  qf[M_ * D_];     // f16 qf tile                  (32 KB)
};

__global__ void __launch_bounds__(128)
gqa_main_kernel(const float* __restrict__ center,
                const float* __restrict__ aff,
                const float* __restrict__ mot,
                const float* __restrict__ keep,
                const float* __restrict__ Wg,
                const float* __restrict__ bg,
                const float* __restrict__ bgbias,
                const _Float16* __restrict__ qf_h,
                const float* __restrict__ qmeta,
                float* __restrict__ out) {
    __shared__ __align__(16) SmemAB sAB;                 // 32 KB (aliased)
    __shared__ __align__(16) _Float16 sGF[ITILE * D_];   // 16 KB normalized gf
    __shared__ float sQM[M_ * 8];                        //  4 KB query meta
    __shared__ float sCI[ITILE * 4];                     //  1 KB {cx,cy,cz,0.5*keeplogit}

    const int tid  = threadIdx.x;
    const int bt   = blockIdx.y;
    const int n0   = blockIdx.x * ITILE;
    const int wv   = tid >> 5;          // wave 0..3
    const int lane = tid & 31;
    const int lr   = lane & 15;         // row/col within a 16-tile
    const int hi   = lane >> 4;         // half-wave select
    const int klo  = hi * 8;            // K chunk offset per 16-bit layout
    const int i0w  = wv * 16;           // wave's row-tile base (local)

    // ---- Phase 0: stage everything into LDS -------------------------------
    for (int idx = tid; idx < D_ * GP_; idx += 128) {
        int d = idx >> 6, k = idx & 63;
        sAB.p1.wg[idx] = (k < G_) ? (_Float16)Wg[(size_t)d * G_ + k] : (_Float16)0.0f;
    }
    for (int idx = tid; idx < ITILE * GP_; idx += 128) {
        int i = idx >> 6, k = idx & 63;
        int n = n0 + i; if (n > N_ - 1) n = N_ - 1;
        float v;
        if (k < 32)      v = aff[((size_t)bt * N_ + n) * 32 + k];
        else if (k < G_) v = mot[((size_t)bt * N_ + n) * 16 + (k - 32)];
        else             v = 0.0f;
        sAB.p1.g[idx] = (_Float16)v;
    }
    if (tid < ITILE) {
        int n = n0 + tid; if (n > N_ - 1) n = N_ - 1;
        const float* c = center + ((size_t)bt * N_ + n) * 3;
        sCI[tid * 4 + 0] = c[0];
        sCI[tid * 4 + 1] = c[1];
        sCI[tid * 4 + 2] = c[2];
        float ks = keep[(size_t)bt * N_ + n];
        ks = fminf(fmaxf(ks, 1e-4f), 1.0f - 1e-4f);
        sCI[tid * 4 + 3] = KEEP_W * (logf(ks) - log1pf(-ks));
    }
    if (tid < M_) {
        const float* qm = qmeta + (size_t)(bt * M_ + tid) * 8;
        #pragma unroll
        for (int j = 0; j < 8; ++j) sQM[tid * 8 + j] = qm[j];
    }
    __syncthreads();

    // ---- Phase 1: gf = gfeat @ Wg.T  (K = 64 padded, 2 WMMA K-steps) ------
    v8f accg[8];
    #pragma unroll
    for (int t = 0; t < 8; ++t) accg[t] = (v8f)(0.0f);

    #pragma unroll
    for (int ks2 = 0; ks2 < 2; ++ks2) {
        const int kbase = 32 * ks2;
        v16h aF = load_frag16(&sAB.p1.g[(i0w + lr) * GP_], kbase, klo);
        #pragma unroll
        for (int t = 0; t < 8; ++t) {
            v16h bF = load_frag16(&sAB.p1.wg[(t * 16 + lr) * GP_], kbase, klo);
            accg[t] = __builtin_amdgcn_wmma_f32_16x16x32_f16(
                false, aF, false, bF, (short)0, accg[t], false, false);
        }
    }
    // + bg, then l2-normalize rows (row r+8*hi lives in half-wave `hi`)
    {
        float bgv[8];
        #pragma unroll
        for (int t = 0; t < 8; ++t) bgv[t] = bg[t * 16 + lr];
        #pragma unroll
        for (int r = 0; r < 8; ++r) {
            float s2 = 0.0f;
            #pragma unroll
            for (int t = 0; t < 8; ++t) {
                float v = accg[t][r] + bgv[t];
                accg[t][r] = v;
                s2 += v * v;
            }
            s2 += __shfl_xor(s2, 1);
            s2 += __shfl_xor(s2, 2);
            s2 += __shfl_xor(s2, 4);
            s2 += __shfl_xor(s2, 8);
            float scl = 1.0f / fmaxf(sqrtf(s2), 1e-12f);
            int row = i0w + r + 8 * hi;
            #pragma unroll
            for (int t = 0; t < 8; ++t)
                sGF[row * D_ + t * 16 + lr] = (_Float16)(accg[t][r] * scl);
        }
    }
    __syncthreads();   // gf written; phase-1 LDS reads complete

    // ---- Phase 2: async-stage the f16 qf tile into the aliased region -----
    // global_load_async_to_lds_b128 (GVS mode): copies memory at saddr + vaddr to LDS at vdst,
    // no VGPR data round-trip; tracked by ASYNCcnt.
    {
        unsigned long long sa = (unsigned long long)(qf_h + (size_t)bt * M_ * D_);
        unsigned int ldsa = (unsigned int)(size_t)(&sAB.qf[0]) + (unsigned int)tid * 16u;
        unsigned int va   = (unsigned int)tid * 16u;
        #pragma unroll
        for (int k = 0; k < 16; ++k) {
            asm volatile("global_load_async_to_lds_b128 %0, %1, %2 offset:0"
                         :: "v"(ldsa), "v"(va), "s"(sa) : "memory");
            ldsa += 2048u;   // 128 threads * 16B per round
            va   += 2048u;
        }
        asm volatile("s_wait_asynccnt 0x0" ::: "memory");
    }
    __syncthreads();

    // ---- Phase 3: sim = gf @ qf.T  (K = 128, 4 WMMA K-steps, 8 N-tiles) ---
    v8f acc[8];
    #pragma unroll
    for (int t = 0; t < 8; ++t) acc[t] = (v8f)(0.0f);

    #pragma unroll
    for (int ks4 = 0; ks4 < 4; ++ks4) {
        const int kbase = 32 * ks4;
        v16h aF = load_frag16(&sGF[(i0w + lr) * D_], kbase, klo);
        #pragma unroll
        for (int t = 0; t < 8; ++t) {
            v16h bF = load_frag16(&sAB.qf[(t * 16 + lr) * D_], kbase, klo);
            acc[t] = __builtin_amdgcn_wmma_f32_16x16x32_f16(
                false, aF, false, bF, (short)0, acc[t], false, false);
        }
    }

    // ---- Phase 4: fused gate + logits + 129-way softmax + store -----------
    float cx8[8], cy8[8], cz8[8], kl8[8];
    #pragma unroll
    for (int r = 0; r < 8; ++r) {
        int il = i0w + r + 8 * hi;
        cx8[r] = sCI[il * 4 + 0];
        cy8[r] = sCI[il * 4 + 1];
        cz8[r] = sCI[il * 4 + 2];
        kl8[r] = sCI[il * 4 + 3];
    }
    const float bgb = bgbias[0];

    #pragma unroll
    for (int t = 0; t < 8; ++t) {
        const float* qm = &sQM[(t * 16 + lr) * 8];
        float qcx = qm[0], qcy = qm[1], qcz = qm[2];
        float qix = qm[3], qiy = qm[4], qiz = qm[5];
        float qpr = qm[6];
        #pragma unroll
        for (int r = 0; r < 8; ++r) {
            float dx = fabsf(cx8[r] - qcx) * qix;
            float dy = fabsf(cy8[r] - qcy) * qiy;
            float dz = fabsf(cz8[r] - qcz) * qiz;
            float sd = fmaxf(dx, fmaxf(dy, dz)) - 1.0f;
            float gate = fminf(fmaxf(-sd * GATE_TEMP, -8.0f), 8.0f);
            acc[t][r] = qpr + FEAT_TEMP * acc[t][r] + gate + kl8[r];
        }
    }

    #pragma unroll
    for (int r = 0; r < 8; ++r) {
        float mx = acc[0][r];
        #pragma unroll
        for (int t = 1; t < 8; ++t) mx = fmaxf(mx, acc[t][r]);
        mx = fmaxf(mx, __shfl_xor(mx, 1));
        mx = fmaxf(mx, __shfl_xor(mx, 2));
        mx = fmaxf(mx, __shfl_xor(mx, 4));
        mx = fmaxf(mx, __shfl_xor(mx, 8));
        mx = fmaxf(mx, bgb);

        float sum = 0.0f;
        #pragma unroll
        for (int t = 0; t < 8; ++t) {
            float e = __expf(acc[t][r] - mx);
            acc[t][r] = e;
            sum += e;
        }
        sum += __shfl_xor(sum, 1);
        sum += __shfl_xor(sum, 2);
        sum += __shfl_xor(sum, 4);
        sum += __shfl_xor(sum, 8);

        float pbg = __expf(bgb - mx);
        float inv = 1.0f / (sum + pbg);

        int i = n0 + i0w + r + 8 * hi;
        if (i < N_) {
            size_t rowoff = ((size_t)bt * N_ + i) * (size_t)(M_ + 1);
            if (lr == 0) out[rowoff] = pbg * inv;
            #pragma unroll
            for (int t = 0; t < 8; ++t)
                out[rowoff + 1 + t * 16 + lr] = acc[t][r] * inv;
        }
    }
}

// ---------------------------------------------------------------------------
extern "C" void kernel_launch(void* const* d_in, const int* in_sizes, int n_in,
                              void* d_out, int out_size, void* d_ws, size_t ws_size,
                              hipStream_t stream) {
    const float* center  = (const float*)d_in[0];
    const float* aff     = (const float*)d_in[1];
    const float* mot     = (const float*)d_in[2];
    const float* keep    = (const float*)d_in[3];
    const float* qbox    = (const float*)d_in[4];
    const float* qfeat   = (const float*)d_in[5];
    const float* pres    = (const float*)d_in[6];
    const float* Wq      = (const float*)d_in[7];
    const float* bq      = (const float*)d_in[8];
    const float* Wg      = (const float*)d_in[9];
    const float* bg      = (const float*)d_in[10];
    const float* bgbias  = (const float*)d_in[11];
    float* out = (float*)d_out;

    // workspace layout
    char* ws = (char*)d_ws;
    _Float16* qf_h  = (_Float16*)ws;                      // 8*128*128 f16 = 256 KB
    float*    qmeta = (float*)(ws + 262144);              // 8*128*8  f32 =  32 KB
    float*    qfraw = (float*)(ws + 262144 + 32768);      // 8*128*128 f32 = 512 KB

    gqa_prep_kernel<<<BT_, 256, 0, stream>>>(qbox, qfeat, pres, Wq, bq,
                                             qf_h, qmeta, qfraw);

    dim3 grid(NTILES, BT_);
    gqa_main_kernel<<<grid, 128, 0, stream>>>(center, aff, mot, keep, Wg, bg,
                                              bgbias, qf_h, qmeta, out);
}
